// VisionAttention_56659208569114
// MI455X (gfx1250) — matrix operand order repaired and checked by reading, not compiled, not run on hardware
//
#include <hip/hip_runtime.h>
#include <hip/hip_bf16.h>
#include <cstdint>

// ---------------------------------------------------------------------------
// MI455X (gfx1250) vision attention:
//   qkv = hs @ Wqkv + b ; RoPE(q,k) ; flash-attn(softmax over valid keys) ;
//   out = ctx @ Wproj + b
// Matmuls: v_wmma_f32_16x16x32_bf16 (f32 accumulate).
// K-block staging in attention: global_load_async_to_lds_b128.
// ---------------------------------------------------------------------------

typedef __bf16 bf16_t;
typedef __attribute__((ext_vector_type(16))) __bf16 v16bf;
typedef __attribute__((ext_vector_type(8)))  __bf16 v8bf;
typedef __attribute__((ext_vector_type(8)))  float  v8f;
typedef __attribute__((ext_vector_type(4)))  int    v4i;

typedef __attribute__((address_space(1))) v4i* as1_v4i_ptr;   // global int4*
typedef __attribute__((address_space(3))) v4i* as3_v4i_ptr;   // LDS int4*

#define NV 4
#define NP 2048
#define ND 1280
#define NH 16
#define HD 80
#define HDP 96     // hd padded to 3*32 for the score GEMM K-dim
#define N3 3840

__device__ __forceinline__ unsigned f2bf_bits(float x) {
  unsigned u = __float_as_uint(x);
  return (u + 0x7FFFu + ((u >> 16) & 1u)) >> 16;   // RNE f32->bf16
}
__device__ __forceinline__ unsigned pack_bf(float lo, float hi) {
  return f2bf_bits(lo) | (f2bf_bits(hi) << 16);
}

__device__ __forceinline__ void wait_async0() {
#if __has_builtin(__builtin_amdgcn_s_wait_asynccnt)
  __builtin_amdgcn_s_wait_asynccnt(0);
#else
  asm volatile("s_wait_asynccnt 0x0" ::: "memory");
#endif
}

// ===========================================================================
// Kernel 1: QKV GEMM (M=8192, N=3840, K=1280), bias, scatter to head-major
// WG tile 128x128, 8 waves (2 M x 4 N), wave tile 64x32 -> 8 WMMA / k-step
// ===========================================================================
__global__ __launch_bounds__(256) void qkv_gemm_kernel(
    const float* __restrict__ hs, const float* __restrict__ w,
    const float* __restrict__ bias,
    bf16_t* __restrict__ qw, bf16_t* __restrict__ kw, bf16_t* __restrict__ vw)
{
  __shared__ alignas(16) bf16_t As[128][32];   // 8 KB
  __shared__ alignas(16) bf16_t Bt[128][32];   // 8 KB, n-major (transposed)

  const int tid  = threadIdx.x;
  const int lane = tid & 31;
  const int wave = tid >> 5;
  const int wm2  = wave >> 2;          // 0..1  (64 M-rows each)
  const int wn4  = wave & 3;           // 0..3  (32 N-cols each)
  const int l15  = lane & 15;
  const int lh   = lane >> 4;
  const int mbase = blockIdx.x * 128;
  const int nbase = blockIdx.y * 128;

  v8f acc[4][2];
#pragma unroll
  for (int mt = 0; mt < 4; ++mt)
#pragma unroll
    for (int nt = 0; nt < 2; ++nt) acc[mt][nt] = (v8f){0,0,0,0,0,0,0,0};

  for (int k0 = 0; k0 < ND; k0 += 32) {
    { // stage A: 128x32 f32 -> bf16, 16 elems/thread, vector LDS writes
      int r = tid >> 1, c = (tid & 1) * 16;
      const float* src = hs + (size_t)(mbase + r) * ND + k0 + c;
      if (k0 + 32 < ND) __builtin_prefetch(src + 32, 0, 1);
      v8bf t0, t1;
#pragma unroll
      for (int i = 0; i < 8; ++i) t0[i] = (bf16_t)src[i];
#pragma unroll
      for (int i = 0; i < 8; ++i) t1[i] = (bf16_t)src[8 + i];
      *(v8bf*)&As[r][c]     = t0;
      *(v8bf*)&As[r][c + 8] = t1;
    }
    { // stage B transposed: thread owns column n, 16 k-values, vector write
      int n = tid >> 1, kk0 = (tid & 1) * 16;
      const float* src = w + (size_t)(k0 + kk0) * N3 + nbase + n;
      v8bf t0, t1;
#pragma unroll
      for (int i = 0; i < 8; ++i) t0[i] = (bf16_t)src[(size_t)i * N3];
#pragma unroll
      for (int i = 0; i < 8; ++i) t1[i] = (bf16_t)src[(size_t)(8 + i) * N3];
      *(v8bf*)&Bt[n][kk0]     = t0;
      *(v8bf*)&Bt[n][kk0 + 8] = t1;
    }
    __syncthreads();

    // A fragments: lane=row, K = {koff..koff+15}, koff=lh*16
    v16bf af[4];
#pragma unroll
    for (int mt = 0; mt < 4; ++mt) {
      union { v16bf v; v8bf h[2]; } ua;
      const bf16_t* ar = &As[wm2 * 64 + mt * 16 + l15][lh * 16];
      ua.h[0] = *(const v8bf*)ar;
      ua.h[1] = *(const v8bf*)(ar + 8);
      af[mt] = ua.v;
    }
#pragma unroll
    for (int nt = 0; nt < 2; ++nt) {
      union { v16bf v; v8bf h[2]; } ub;   // lane=column, 16 contiguous K
      const bf16_t* br = &Bt[wn4 * 32 + nt * 16 + l15][lh * 16];
      ub.h[0] = *(const v8bf*)br;
      ub.h[1] = *(const v8bf*)(br + 8);
#pragma unroll
      for (int mt = 0; mt < 4; ++mt)
        acc[mt][nt] = __builtin_amdgcn_wmma_f32_16x16x32_bf16(
            false, af[mt], false, ub.v, (short)0, acc[mt][nt], false, false);
    }
    __syncthreads();
  }

  // epilogue: bias add + scatter to q/k (hd stride 96) or v (hd stride 80)
#pragma unroll
  for (int nt = 0; nt < 2; ++nt) {
    int n = nbase + wn4 * 32 + nt * 16 + l15;
    float bv = bias[n];
    int sel = n / ND;            // 0=q 1=k 2=v
    int d   = n - sel * ND;
    int h   = d / HD;
    int dd  = d - h * HD;
#pragma unroll
    for (int mt = 0; mt < 4; ++mt) {
      int m0 = mbase + wm2 * 64 + mt * 16 + lh * 8;
#pragma unroll
      for (int i = 0; i < 8; ++i) {
        int m = m0 + i;
        int v = m >> 11;
        int p = m & (NP - 1);
        float val = acc[mt][nt][i] + bv;
        size_t row = (size_t)(v * NH + h) * NP + p;
        if (sel == 0)      qw[row * HDP + dd] = (bf16_t)val;
        else if (sel == 1) kw[row * HDP + dd] = (bf16_t)val;
        else               vw[row * HD  + dd] = (bf16_t)val;
      }
    }
  }
}

// ===========================================================================
// Kernel 2: RoPE on q,k (in place, pair (d, d+40)) + zero hd padding 80..95
// ===========================================================================
__global__ __launch_bounds__(64) void rope_kernel(
    bf16_t* __restrict__ qw, bf16_t* __restrict__ kw,
    const float* __restrict__ rc)     // [V][P][160]  cos|sin
{
  int row = blockIdx.x;               // (v*16+h)*2048 + p
  int p   = row & (NP - 1);
  int v   = (row >> 11) >> 4;
  int t   = threadIdx.x;
  bf16_t* qr = qw + (size_t)row * HDP;
  bf16_t* kr = kw + (size_t)row * HDP;
  if (t < 40) {
    const float* rcp = rc + ((size_t)v * NP + p) * (2 * HD);
    float c0 = rcp[t],      c1 = rcp[t + 40];
    float s0 = rcp[HD + t], s1 = rcp[HD + t + 40];
    float q0 = (float)qr[t], q1 = (float)qr[t + 40];
    qr[t]      = (bf16_t)(q0 * c0 - q1 * s0);   // rh(x)[i<40] = -x[i+40]
    qr[t + 40] = (bf16_t)(q1 * c1 + q0 * s1);   // rh(x)[i>=40] = x[i-40]
    float k0 = (float)kr[t], k1 = (float)kr[t + 40];
    kr[t]      = (bf16_t)(k0 * c0 - k1 * s0);
    kr[t + 40] = (bf16_t)(k1 * c1 + k0 * s1);
  } else if (t < 56) {
    qr[HD + (t - 40)] = (bf16_t)0.0f;
    kr[HD + (t - 40)] = (bf16_t)0.0f;
  }
}

// ===========================================================================
// Kernel 3: flash attention, one workgroup = 128 q-rows of one (v,h)
// S^T = K x Q^T ; online softmax per lane (+ shfl_xor(16) partner)
// O^T = V^T x P^T (5 hd-tiles of 16 = 80)
// K block staged via async-to-LDS DMA.
// ===========================================================================
__global__ __launch_bounds__(256) void attn_kernel(
    const bf16_t* __restrict__ qw, const bf16_t* __restrict__ kw,
    const bf16_t* __restrict__ vw, const int* __restrict__ valid_lens,
    bf16_t* __restrict__ ctx)
{
  __shared__ alignas(16) bf16_t Kt[32][HDP];   // key-major K block (6 KB)
  __shared__ alignas(16) bf16_t Vt[HD][32];    // hd-major V^T block (5 KB)

  const int tid  = threadIdx.x;
  const int lane = tid & 31;
  const int wave = tid >> 5;
  const int l15  = lane & 15;
  const int lh   = lane >> 4;

  const int blk = blockIdx.x;          // (v*16+h)*16 + qblock
  const int qb  = blk & 15;
  const int vh  = blk >> 4;
  const int v   = vh >> 4;
  const int h   = vh & 15;

  const bf16_t* qbase = qw + (size_t)vh * NP * HDP;
  const bf16_t* kbase = kw + (size_t)vh * NP * HDP;
  const bf16_t* vbase = vw + (size_t)vh * NP * HD;

  const int qrow0 = qb * 128 + wave * 16;

  // resident Q^T B-fragments for the 3 K-windows of HDP=96
  v16bf qf[3];
  {
    const bf16_t* qr = qbase + (size_t)(qrow0 + l15) * HDP + lh * 8;
#pragma unroll
    for (int wdw = 0; wdw < 3; ++wdw) {
      union { v16bf v; v8bf h2[2]; } u;
      u.h2[0] = *(const v8bf*)(qr + wdw * 32);
      u.h2[1] = *(const v8bf*)(qr + wdw * 32 + 16);
      qf[wdw] = u.v;
    }
  }

  int vlen = valid_lens[v];
  if (vlen < 1) vlen = 1;
  if (vlen > NP) vlen = NP;
  const int nkb = (vlen + 31) >> 5;

  v8f o[5];
#pragma unroll
  for (int t = 0; t < 5; ++t) o[t] = (v8f){0,0,0,0,0,0,0,0};
  float m_run = -1e30f, s_run = 0.0f;
  const float scale = 0.1118033988749895f;   // 80^-0.5

  for (int kb = 0; kb < nkb; ++kb) {
    { // ---- stage K block 32x96 bf16 (6144 B) via async DMA to LDS ----
      char* srcb = (char*)(kbase + (size_t)kb * 32 * HDP);
      char* dstb = (char*)&Kt[0][0];
#if __has_builtin(__builtin_amdgcn_global_load_async_to_lds_b128)
      __builtin_amdgcn_global_load_async_to_lds_b128(
          (as1_v4i_ptr)(srcb + tid * 16),
          (as3_v4i_ptr)(dstb + tid * 16), 0, 0);
      if (tid < 128)
        __builtin_amdgcn_global_load_async_to_lds_b128(
            (as1_v4i_ptr)(srcb + 4096 + tid * 16),
            (as3_v4i_ptr)(dstb + 4096 + tid * 16), 0, 0);
#else
      const unsigned* src = (const unsigned*)srcb;
      unsigned* dst = (unsigned*)dstb;
#pragma unroll
      for (int i = 0; i < 6; ++i) dst[tid + i * 256] = src[tid + i * 256];
#endif
    }
    { // ---- stage V block transposed: Vt[dd][key] ----
      const bf16_t* src = vbase + (size_t)kb * 32 * HD;
#pragma unroll
      for (int i = 0; i < 10; ++i) {
        int idx = tid + i * 256;          // 0..2559
        int key = idx / HD;
        int dd  = idx - key * HD;
        Vt[dd][key] = src[idx];
      }
    }
#if __has_builtin(__builtin_amdgcn_global_load_async_to_lds_b128)
    wait_async0();
#endif
    __syncthreads();

    // --- S^T tiles (M=key, N=qrow): 2 key-tiles x 3 k-windows = 6 WMMA ---
    v8f sc[2];
#pragma unroll
    for (int t = 0; t < 2; ++t) {
      v8f c = {0,0,0,0,0,0,0,0};
#pragma unroll
      for (int wdw = 0; wdw < 3; ++wdw) {
        union { v16bf v; v8bf h2[2]; } ua;
        const bf16_t* krow = &Kt[t * 16 + l15][wdw * 32 + lh * 8];
        ua.h2[0] = *(const v8bf*)krow;
        ua.h2[1] = *(const v8bf*)(krow + 16);
        c = __builtin_amdgcn_wmma_f32_16x16x32_bf16(
            false, ua.v, false, qf[wdw], (short)0, c, false, false);
      }
      sc[t] = c;
    }

    // --- scale + mask + online softmax ---
    float vals[2][8];
    float mloc = -1e30f;
#pragma unroll
    for (int t = 0; t < 2; ++t) {
      int keyb = kb * 32 + t * 16 + lh * 8;
#pragma unroll
      for (int j = 0; j < 8; ++j) {
        float x = sc[t][j] * scale;
        if (keyb + j >= vlen) x = -1e9f;     // NEG_INF mask
        vals[t][j] = x;
        mloc = fmaxf(mloc, x);
      }
    }
    mloc = fmaxf(mloc, __shfl_xor(mloc, 16, 32));
    float m_new = fmaxf(m_run, mloc);
    float corr  = __expf(m_run - m_new);
    float lsum  = 0.0f;
    unsigned pk[2][4];
#pragma unroll
    for (int t = 0; t < 2; ++t) {
      float pr[8];
#pragma unroll
      for (int j = 0; j < 8; ++j) { pr[j] = __expf(vals[t][j] - m_new); lsum += pr[j]; }
#pragma unroll
      for (int j = 0; j < 4; ++j) pk[t][j] = pack_bf(pr[2 * j], pr[2 * j + 1]);
    }
    lsum += __shfl_xor(lsum, 16, 32);
    s_run = s_run * corr + lsum;
    m_run = m_new;
#pragma unroll
    for (int t = 0; t < 5; ++t)
#pragma unroll
      for (int j = 0; j < 8; ++j) o[t][j] *= corr;

    // --- build P^T B-fragment (32 keys) via lane^16 exchange ---
    union { v16bf v; unsigned u[8]; } pb;
#pragma unroll
    for (int j = 0; j < 4; ++j) {
      unsigned o0 = (unsigned)__shfl_xor((int)pk[0][j], 16, 32);
      unsigned o1 = (unsigned)__shfl_xor((int)pk[1][j], 16, 32);
      pb.u[j]     = (lane < 16) ? pk[0][j] : o1;
      pb.u[4 + j] = (lane < 16) ? o0       : pk[1][j];
    }

    // --- O^T += V^T x P^T : 5 WMMA ---
#pragma unroll
    for (int t = 0; t < 5; ++t) {
      union { v16bf v; v8bf h2[2]; } ua;
      const bf16_t* vr = &Vt[t * 16 + l15][lh * 16];
      ua.h2[0] = *(const v8bf*)vr;
      ua.h2[1] = *(const v8bf*)(vr + 8);
      o[t] = __builtin_amdgcn_wmma_f32_16x16x32_bf16(
          false, ua.v, false, pb.v, (short)0, o[t], false, false);
    }
    __syncthreads();
  }

  // finalize: divide by row-sum, store context bf16 [v][p][h*80+dd]
  float inv = 1.0f / s_run;
  int p = qrow0 + l15;
  bf16_t* crow = ctx + ((size_t)v * NP + p) * ND + h * HD;
#pragma unroll
  for (int t = 0; t < 5; ++t) {
    uint4 wv;
    wv.x = pack_bf(o[t][0] * inv, o[t][1] * inv);
    wv.y = pack_bf(o[t][2] * inv, o[t][3] * inv);
    wv.z = pack_bf(o[t][4] * inv, o[t][5] * inv);
    wv.w = pack_bf(o[t][6] * inv, o[t][7] * inv);
    *(uint4*)(crow + t * 16 + lh * 8) = wv;   // dd = t*16 + lh*8 + j
  }
}

// ===========================================================================
// Kernel 4: output projection GEMM (M=8192, N=1280, K=1280), f32 out + bias
// same 128x128 tiling as kernel 1, A already bf16
// ===========================================================================
__global__ __launch_bounds__(256) void proj_gemm_kernel(
    const bf16_t* __restrict__ ctx, const float* __restrict__ w,
    const float* __restrict__ bias, float* __restrict__ out)
{
  __shared__ alignas(16) bf16_t As[128][32];
  __shared__ alignas(16) bf16_t Bt[128][32];

  const int tid  = threadIdx.x;
  const int lane = tid & 31;
  const int wave = tid >> 5;
  const int wm2  = wave >> 2;
  const int wn4  = wave & 3;
  const int l15  = lane & 15;
  const int lh   = lane >> 4;
  const int mbase = blockIdx.x * 128;
  const int nbase = blockIdx.y * 128;

  v8f acc[4][2];
#pragma unroll
  for (int mt = 0; mt < 4; ++mt)
#pragma unroll
    for (int nt = 0; nt < 2; ++nt) acc[mt][nt] = (v8f){0,0,0,0,0,0,0,0};

  for (int k0 = 0; k0 < ND; k0 += 32) {
    { // stage A (bf16 source, 32 B per thread)
      int r = tid >> 1, c = (tid & 1) * 16;
      const bf16_t* src = ctx + (size_t)(mbase + r) * ND + k0 + c;
      if (k0 + 32 < ND) __builtin_prefetch(src + 32, 0, 1);
      *(v8bf*)&As[r][c]     = *(const v8bf*)src;
      *(v8bf*)&As[r][c + 8] = *(const v8bf*)(src + 8);
    }
    { // stage B transposed (f32 -> bf16)
      int n = tid >> 1, kk0 = (tid & 1) * 16;
      const float* src = w + (size_t)(k0 + kk0) * ND + nbase + n;
      v8bf t0, t1;
#pragma unroll
      for (int i = 0; i < 8; ++i) t0[i] = (bf16_t)src[(size_t)i * ND];
#pragma unroll
      for (int i = 0; i < 8; ++i) t1[i] = (bf16_t)src[(size_t)(8 + i) * ND];
      *(v8bf*)&Bt[n][kk0]     = t0;
      *(v8bf*)&Bt[n][kk0 + 8] = t1;
    }
    __syncthreads();

    v16bf af[4];
#pragma unroll
    for (int mt = 0; mt < 4; ++mt) {
      union { v16bf v; v8bf h[2]; } ua;
      const bf16_t* ar = &As[wm2 * 64 + mt * 16 + l15][lh * 16];
      ua.h[0] = *(const v8bf*)ar;
      ua.h[1] = *(const v8bf*)(ar + 8);
      af[mt] = ua.v;
    }
#pragma unroll
    for (int nt = 0; nt < 2; ++nt) {
      union { v16bf v; v8bf h[2]; } ub;
      const bf16_t* br = &Bt[wn4 * 32 + nt * 16 + l15][lh * 16];
      ub.h[0] = *(const v8bf*)br;
      ub.h[1] = *(const v8bf*)(br + 8);
#pragma unroll
      for (int mt = 0; mt < 4; ++mt)
        acc[mt][nt] = __builtin_amdgcn_wmma_f32_16x16x32_bf16(
            false, af[mt], false, ub.v, (short)0, acc[mt][nt], false, false);
    }
    __syncthreads();
  }

#pragma unroll
  for (int nt = 0; nt < 2; ++nt) {
    int n = nbase + wn4 * 32 + nt * 16 + l15;
    float bv = bias[n];
#pragma unroll
    for (int mt = 0; mt < 4; ++mt) {
      int m0 = mbase + wm2 * 64 + mt * 16 + lh * 8;
#pragma unroll
      for (int i = 0; i < 8; ++i)
        out[(size_t)(m0 + i) * ND + n] = acc[mt][nt][i] + bv;
    }
  }
}

// ===========================================================================
extern "C" void kernel_launch(void* const* d_in, const int* in_sizes, int n_in,
                              void* d_out, int out_size, void* d_ws, size_t ws_size,
                              hipStream_t stream) {
  (void)in_sizes; (void)n_in; (void)out_size; (void)ws_size;
  const float* hs    = (const float*)d_in[0];   // (4,2048,1280)
  const float* rc    = (const float*)d_in[1];   // (4,2048,1,160)
  const int*   vlens = (const int*)  d_in[2];   // (4,)
  const float* wqkv  = (const float*)d_in[3];   // (1280,3840)
  const float* bqkv  = (const float*)d_in[4];   // (3840,)
  const float* wproj = (const float*)d_in[5];   // (1280,1280)
  const float* bproj = (const float*)d_in[6];   // (1280,)
  // d_in[7] = n_heads (16), hardcoded

  const size_t qk_elems = (size_t)NV * NH * NP * HDP;   // 12,582,912
  const size_t v_elems  = (size_t)NV * NH * NP * HD;    // 10,485,760
  bf16_t* qw  = (bf16_t*)d_ws;
  bf16_t* kw  = qw + qk_elems;
  bf16_t* vw  = kw + qk_elems;
  bf16_t* ctx = vw + v_elems;                           // V*P*D bf16

  float* out = (float*)d_out;

  qkv_gemm_kernel<<<dim3((NV * NP) / 128, N3 / 128), 256, 0, stream>>>(
      hs, wqkv, bqkv, qw, kw, vw);
  rope_kernel<<<dim3(NV * NH * NP), 64, 0, stream>>>(qw, kw, rc);
  attn_kernel<<<dim3(NV * NH * (NP / 128)), 256, 0, stream>>>(
      qw, kw, vw, vlens, ctx);
  proj_gemm_kernel<<<dim3((NV * NP) / 128, ND / 128), 256, 0, stream>>>(
      ctx, wproj, bproj, out);
}
